// MyModel_78872779424027
// MI455X (gfx1250) — matrix operand order repaired and checked by reading, not compiled, not run on hardware
//
#include <hip/hip_runtime.h>
#include <math.h>

// Problem constants (match reference)
#define BB 16
#define NN 2048
#define EE 64
#define FF 8
#define TT 5

typedef __attribute__((ext_vector_type(16))) __bf16 v16bf;
typedef __attribute__((ext_vector_type(8)))  float  v8f;

union BF16x16 { v16bf v; unsigned short u[16]; uint4 q[2]; };
union F32x16  { float4 f4[4]; float f[16]; };

// split f32 into bf16 hi (truncate) and bf16 lo (truncate of exact residual)
__device__ __forceinline__ void split_bf16(float x, unsigned short& hi, unsigned short& lo) {
    unsigned int u = __float_as_uint(x);
    hi = (unsigned short)(u >> 16);
    float r = x - __uint_as_float(u & 0xFFFF0000u);   // exact residual
    lo = (unsigned short)(__float_as_uint(r) >> 16);
}

// ---------------------------------------------------------------------------
// Kernel A: term[g][b][e][n] = sum_f W1[e][f] * feat[b][n][f]
// ---------------------------------------------------------------------------
__global__ void term_kernel(const float* __restrict__ feat1,
                            const float* __restrict__ feat2,
                            const float* __restrict__ W1,
                            float* __restrict__ term) {
    size_t idx = (size_t)blockIdx.x * blockDim.x + threadIdx.x;  // < 2*16*64*2048
    int n  = (int)(idx % NN);
    int e  = (int)((idx / NN) % EE);
    int b  = (int)((idx / ((size_t)NN * EE)) % BB);
    int gg = (int)(idx / ((size_t)NN * EE * BB));
    const float* feat = gg ? feat2 : feat1;
    const float* fr = feat + ((size_t)b * NN + n) * FF;
    const float* wr = W1 + e * FF;
    float s = 0.f;
#pragma unroll
    for (int f = 0; f < FF; ++f) s += wr[f] * fr[f];
    term[idx] = s;
}

// ---------------------------------------------------------------------------
// Kernel B: per column n:  h = theta1 @ relu(theta @ prev_col)  (h=0 if first)
//           c = tanh(term + h); write curr f32 and bf16 hi/lo splits
// grid (8,16,2) x 256 threads; column = blockIdx.x*256 + tid
// ---------------------------------------------------------------------------
__global__ void __launch_bounds__(256)
embed_kernel(const float* __restrict__ prev, const float* __restrict__ term,
             const float* __restrict__ theta, const float* __restrict__ theta1,
             float* __restrict__ curr,
             unsigned short* __restrict__ chi, unsigned short* __restrict__ clo,
             int first) {
    __shared__ float th[EE * EE];
    __shared__ float th1[EE * EE];
    for (int i = threadIdx.x; i < EE * EE; i += blockDim.x) {
        th[i]  = theta[i];
        th1[i] = theta1[i];
    }
    __syncthreads();

    int n  = blockIdx.x * blockDim.x + threadIdx.x;
    int b  = blockIdx.y;
    int gg = blockIdx.z;
    size_t base = (((size_t)gg * BB + b) * EE) * NN + n;

    float h[EE];
    if (first) {
#pragma unroll
        for (int i = 0; i < EE; ++i) h[i] = 0.f;
    } else {
        float v[EE];
#pragma unroll
        for (int j = 0; j < EE; ++j) v[j] = prev[base + (size_t)j * NN];
        float u[EE];
#pragma unroll
        for (int i = 0; i < EE; ++i) {
            float s = 0.f;
#pragma unroll
            for (int j = 0; j < EE; ++j) s += th[i * EE + j] * v[j];
            u[i] = fmaxf(s, 0.f);
        }
#pragma unroll
        for (int i = 0; i < EE; ++i) {
            float s = 0.f;
#pragma unroll
            for (int j = 0; j < EE; ++j) s += th1[i * EE + j] * u[j];
            h[i] = s;
        }
    }
#pragma unroll
    for (int e = 0; e < EE; ++e) {
        size_t p = base + (size_t)e * NN;
        float c = tanhf(term[p] + h[e]);
        curr[p] = c;
        unsigned short hi, lo;
        split_bf16(c, hi, lo);
        chi[p] = hi;
        clo[p] = lo;
    }
}

// ---------------------------------------------------------------------------
// Kernel C (WMMA): prev[g][b][e][m] = sum_n adj[b][m][n] * curr[g][b][e][n]
// grid (32 m-groups, 16 b, 2 g) x 128 threads (4 waves).
// Each wave owns ONE 16-wide m-tile and computes ALL FOUR 16-row e-tiles:
//   - adj loaded + split to bf16 hi/lo exactly once per element
//   - 12 WMMAs per k-step across 4 independent accumulators (no RAW chains)
// split-bf16 product: hi*hi + hi*lo + lo*hi, f32 accumulate
// ---------------------------------------------------------------------------
__global__ void __launch_bounds__(128)
aggregate_kernel(const unsigned short* __restrict__ chi,
                 const unsigned short* __restrict__ clo,
                 const float* __restrict__ adj1, const float* __restrict__ adj2,
                 float* __restrict__ prev) {
    const int wave = threadIdx.x >> 5;
    const int lane = threadIdx.x & 31;
    const int half = lane >> 4;      // 0: lanes 0-15, 1: lanes 16-31
    const int l16  = lane & 15;
    const int m0 = (blockIdx.x * 4 + wave) * 16;
    const int b  = blockIdx.y;
    const int gg = blockIdx.z;
    const float* adj = gg ? adj2 : adj1;

    // A-frag chunk offsets within a K=32 step (contiguous k = 16*half + s, matches B)
    const int c1 = half ? 16 : 0;
    const int c2 = half ? 24 : 8;

    // A rows for the 4 e-tiles: e = et*16 + l16
    size_t arow[4];
#pragma unroll
    for (int et = 0; et < 4; ++et)
        arow[et] = (((size_t)gg * BB + b) * EE + (et * 16 + l16)) * NN;

    // B-frag source: adj row m = m0 + l16, 16 contiguous n at chunk half*16
    const size_t brow = ((size_t)b * NN + (m0 + l16)) * NN + half * 16;

    v8f acc[4];
#pragma unroll
    for (int et = 0; et < 4; ++et)
        acc[et] = (v8f){0.f, 0.f, 0.f, 0.f, 0.f, 0.f, 0.f, 0.f};

    for (int k0 = 0; k0 < NN; k0 += 32) {
        // ---- B tile: load f32 adj, split to bf16 hi/lo ONCE ----
        F32x16 x;
        const float4* bp = (const float4*)(adj + brow + k0);
        x.f4[0] = bp[0]; x.f4[1] = bp[1]; x.f4[2] = bp[2]; x.f4[3] = bp[3];
        BF16x16 bh, bl;
#pragma unroll
        for (int j = 0; j < 16; ++j) split_bf16(x.f[j], bh.u[j], bl.u[j]);

        // ---- A frags for the 4 e-tiles (pre-split bf16 in workspace) ----
        BF16x16 ah[4], al[4];
#pragma unroll
        for (int et = 0; et < 4; ++et) {
            ah[et].q[0] = *(const uint4*)(chi + arow[et] + k0 + c1);
            ah[et].q[1] = *(const uint4*)(chi + arow[et] + k0 + c2);
            al[et].q[0] = *(const uint4*)(clo + arow[et] + k0 + c1);
            al[et].q[1] = *(const uint4*)(clo + arow[et] + k0 + c2);
        }

        // ---- 12 WMMAs, rotating across 4 independent accumulators ----
#pragma unroll
        for (int et = 0; et < 4; ++et)
            acc[et] = __builtin_amdgcn_wmma_f32_16x16x32_bf16(
                false, ah[et].v, false, bh.v, (short)0, acc[et], false, false);
#pragma unroll
        for (int et = 0; et < 4; ++et)
            acc[et] = __builtin_amdgcn_wmma_f32_16x16x32_bf16(
                false, ah[et].v, false, bl.v, (short)0, acc[et], false, false);
#pragma unroll
        for (int et = 0; et < 4; ++et)
            acc[et] = __builtin_amdgcn_wmma_f32_16x16x32_bf16(
                false, al[et].v, false, bh.v, (short)0, acc[et], false, false);
    }

    // D-frag store: col m = m0 + l16; row e = et*16 + half*8 + r
    const int m = m0 + l16;
#pragma unroll
    for (int et = 0; et < 4; ++et) {
#pragma unroll
        for (int r = 0; r < 8; ++r) {
            int e = et * 16 + half * 8 + r;
            prev[(((size_t)gg * BB + b) * EE + e) * NN + m] = acc[et][r];
        }
    }
}

// ---------------------------------------------------------------------------
// Kernel D: gsum[(g*B+b)*E+e] = sum_n curr[...][n]   (one block per row)
// ---------------------------------------------------------------------------
__global__ void __launch_bounds__(256)
gsum_kernel(const float* __restrict__ curr, float* __restrict__ gsum) {
    __shared__ float red[256];
    size_t base = (size_t)blockIdx.x * NN;
    float s = 0.f;
    for (int n = threadIdx.x; n < NN; n += 256) s += curr[base + n];
    red[threadIdx.x] = s;
    __syncthreads();
    for (int off = 128; off > 0; off >>= 1) {
        if (threadIdx.x < off) red[threadIdx.x] += red[threadIdx.x + off];
        __syncthreads();
    }
    if (threadIdx.x == 0) gsum[blockIdx.x] = red[0];
}

// ---------------------------------------------------------------------------
// Kernel E: e_g = W2 @ g^T  (E x B each graph), cosine sim over E dim
// out layout: sim[16] | e1[64*16] | e2[64*16]
// ---------------------------------------------------------------------------
__global__ void __launch_bounds__(256)
final_kernel(const float* __restrict__ gsum, const float* __restrict__ W2,
             float* __restrict__ out) {
    __shared__ float w[EE * EE];
    __shared__ float gs[2 * BB * EE];
    __shared__ float esh[2 * EE * BB];
    for (int i = threadIdx.x; i < EE * EE; i += 256) w[i] = W2[i];
    for (int i = threadIdx.x; i < 2 * BB * EE; i += 256) gs[i] = gsum[i];
    __syncthreads();
    for (int idx = threadIdx.x; idx < 2 * EE * BB; idx += 256) {
        int gg  = idx >> 10;           // /1024
        int rem = idx & 1023;
        int i   = rem >> 4;            // E index
        int b   = rem & 15;            // B index
        float s = 0.f;
#pragma unroll
        for (int j = 0; j < EE; ++j) s += w[i * EE + j] * gs[gg * (BB * EE) + b * EE + j];
        esh[idx] = s;
        out[16 + idx] = s;             // e1 then e2, row-major [E][B]
    }
    __syncthreads();
    if (threadIdx.x < BB) {
        int b = threadIdx.x;
        float d = 0.f, s1 = 0.f, s2 = 0.f;
#pragma unroll
        for (int i = 0; i < EE; ++i) {
            float a = esh[i * BB + b];
            float c = esh[EE * BB + i * BB + b];
            d += a * c; s1 += a * a; s2 += c * c;
        }
        float n1 = fmaxf(sqrtf(s1), 1e-6f);
        float n2 = fmaxf(sqrtf(s2), 1e-6f);
        out[b] = d / (n1 * n2);
    }
}

// ---------------------------------------------------------------------------
extern "C" void kernel_launch(void* const* d_in, const int* in_sizes, int n_in,
                              void* d_out, int out_size, void* d_ws, size_t ws_size,
                              hipStream_t stream) {
    const float* adj1  = (const float*)d_in[0];
    const float* feat1 = (const float*)d_in[1];
    const float* adj2  = (const float*)d_in[2];
    const float* feat2 = (const float*)d_in[3];
    const float* W1    = (const float*)d_in[4];
    const float* W2    = (const float*)d_in[5];
    const float* theta  = (const float*)d_in[6];
    const float* theta1 = (const float*)d_in[7];
    float* out = (float*)d_out;

    const size_t NEL = (size_t)2 * BB * EE * NN;   // 4M elements per full tensor
    char* ws = (char*)d_ws;
    float*          term = (float*)ws;                               // 16 MB
    float*          prev = term + NEL;                               // 16 MB
    float*          curr = prev + NEL;                               // 16 MB
    unsigned short* chi  = (unsigned short*)(curr + NEL);            //  8 MB
    unsigned short* clo  = chi + NEL;                                //  8 MB
    float*          gsum = (float*)(clo + NEL);                      //  8 KB
    (void)ws_size; (void)in_sizes; (void)n_in; (void)out_size;       // ~64 MB total

    // term = W1 @ feat^T for both graphs
    term_kernel<<<(unsigned)(NEL / 256), 256, 0, stream>>>(feat1, feat2, W1, term);

    // T recurrent steps; last aggregation is dead in the reference, skip it
    for (int t = 0; t < TT; ++t) {
        embed_kernel<<<dim3(NN / 256, BB, 2), 256, 0, stream>>>(
            prev, term, theta, theta1, curr, chi, clo, t == 0 ? 1 : 0);
        if (t < TT - 1) {
            aggregate_kernel<<<dim3(NN / 64, BB, 2), 128, 0, stream>>>(
                chi, clo, adj1, adj2, prev);
        }
    }

    // g = curr.sum(axis=n), then W2 @ g^T and cosine similarity
    gsum_kernel<<<2 * BB * EE, 256, 0, stream>>>(curr, gsum);
    final_kernel<<<1, 256, 0, stream>>>(gsum, W2, out);
}
